// _MambaBlock_21672404976174
// MI455X (gfx1250) — compile-verified, matrix operand order checked by reading, not run on hardware
//
#include <hip/hip_runtime.h>
#include <math.h>

typedef __attribute__((ext_vector_type(16))) _Float16 v16h;
typedef __attribute__((ext_vector_type(8)))  _Float16 v8h;
typedef __attribute__((ext_vector_type(8)))  float    v8f;

#define D_MODEL 1024
#define D_STATE 16
#define D_CONV  4
#define D_INNER 2048
#define DT_RANK 64
#define BATCH   2
#define SEQ     2048
#define NROWS   (BATCH*SEQ)            /* 4096 token rows */
#define DBC_W   (DT_RANK + 2*D_STATE)  /* 96 */
#define LN_EPS  1e-5f

__device__ __forceinline__ float sigmoidf_(float v){ return 1.f/(1.f+__expf(-v)); }

// Load one 16-element f16 WMMA fragment slice for this lane.
// Per ISA layout: lane l (h = l>>4) holds K chunks [8h..8h+7] and [16+8h..16+8h+7];
// caller folds the 8h offset into p, so chunks are at p[0..7] and p[16..23].
__device__ __forceinline__ v16h frag16(const _Float16* __restrict__ p){
  v8h c0 = *(const v8h*)(p);        // 16B aligned
  v8h c1 = *(const v8h*)(p + 16);
  return __builtin_shufflevector(c0, c1, 0,1,2,3,4,5,6,7,8,9,10,11,12,13,14,15);
}

__device__ __forceinline__ v8f wmma_(v16h a, v16h b, v8f c){
  return __builtin_amdgcn_wmma_f32_16x16x32_f16(false, a, false, b,
                                                (short)0, c, false, false);
}

// ---------------------------------------------------------------------------
// LayerNorm: one block per token row; writes f16 output (feeds WMMA GEMM only)
// ---------------------------------------------------------------------------
__global__ void ln_kernel(const float* __restrict__ x, const float* __restrict__ w,
                          const float* __restrict__ b, _Float16* __restrict__ xnh){
  int row = blockIdx.x;
  const float* xr = x + (size_t)row * D_MODEL;
  float v[4];
  float s = 0.f, s2 = 0.f;
#pragma unroll
  for (int i = 0; i < 4; i++){
    v[i] = xr[threadIdx.x + i*256];
    s += v[i]; s2 += v[i]*v[i];
  }
  for (int off = 16; off > 0; off >>= 1){
    s  += __shfl_down(s,  off, 32);
    s2 += __shfl_down(s2, off, 32);
  }
  __shared__ float ps[8], ps2[8];
  __shared__ float mu, rstd;
  int wid = threadIdx.x >> 5;
  if ((threadIdx.x & 31) == 0){ ps[wid] = s; ps2[wid] = s2; }
  __syncthreads();
  if (threadIdx.x == 0){
    float ts = 0.f, ts2 = 0.f;
    for (int i = 0; i < 8; i++){ ts += ps[i]; ts2 += ps2[i]; }
    float m   = ts  / (float)D_MODEL;
    float var = ts2 / (float)D_MODEL - m*m;
    mu = m; rstd = rsqrtf(var + LN_EPS);
  }
  __syncthreads();
  _Float16* xo = xnh + (size_t)row * D_MODEL;
#pragma unroll
  for (int i = 0; i < 4; i++){
    int c = threadIdx.x + i*256;
    xo[c] = (_Float16)((v[i] - mu) * rstd * w[c] + b[c]);
  }
}

// ---------------------------------------------------------------------------
// Generic f32 -> f16 conversion (one pass per weight tensor)
// ---------------------------------------------------------------------------
__global__ void cvt_f16_kernel(const float* __restrict__ src,
                               _Float16* __restrict__ dst, int n){
  int i = blockIdx.x * blockDim.x + threadIdx.x;
  if (i < n) dst[i] = (_Float16)src[i];
}

// Extract dt slice of dbc (cols 0..63) as f16 with packed lda=64
__global__ void dbc_dt_f16_kernel(const float* __restrict__ dbc,
                                  _Float16* __restrict__ dth){
  int i = blockIdx.x * blockDim.x + threadIdx.x;   // NROWS*64
  int m = i >> 6, r = i & 63;
  dth[i] = (_Float16)dbc[(size_t)m*DBC_W + r];
}

// ---------------------------------------------------------------------------
// WMMA GEMM, f16 operands, f32 out: C[M,N] = A[M,K] * Bw[N,K]^T (+ resid).
// One wave computes a 32x32 C tile: 2 A-frags x 2 B-frags -> 4 wmma per K-step.
// ---------------------------------------------------------------------------
__device__ __forceinline__ void store_sub(float* __restrict__ C, int ldc,
                                          const float* __restrict__ resid,
                                          int row0, int col0, v8f a){
  float* cp = C + (size_t)row0 * ldc + col0;
  if (resid){
    const float* rp = resid + (size_t)row0 * ldc + col0;
#pragma unroll
    for (int r = 0; r < 8; r++) cp[(size_t)r*ldc] = a[r] + rp[(size_t)r*ldc];
  } else {
#pragma unroll
    for (int r = 0; r < 8; r++) cp[(size_t)r*ldc] = a[r];
  }
}

__global__ void wmma_gemm32(const _Float16* __restrict__ A, int lda,
                            const _Float16* __restrict__ Bw, int ldb,
                            float* __restrict__ C, int ldc,
                            int M, int N, int K,
                            const float* __restrict__ resid){
  const int lane   = threadIdx.x & 31;
  const int wave   = threadIdx.x >> 5;
  const int tilesN = N >> 5;
  const int numTiles = (M >> 5) * tilesN;
  int tile = blockIdx.x * (blockDim.x >> 5) + wave;   // uniform per wave
  if (tile >= numTiles) return;                        // whole-wave exit
  const int tm   = tile / tilesN;
  const int tn   = tile - tm * tilesN;
  const int half = lane >> 4;
  const int r16  = lane & 15;

  const _Float16* a0 = A  + (size_t)(tm*32 + r16) * lda + half*8;
  const _Float16* a1 = a0 + (size_t)16 * lda;
  const _Float16* b0 = Bw + (size_t)(tn*32 + r16) * ldb + half*8;
  const _Float16* b1 = b0 + (size_t)16 * ldb;

  v8f acc00 = {}, acc01 = {}, acc10 = {}, acc11 = {};
  for (int k0 = 0; k0 < K; k0 += 32){
    v16h va0 = frag16(a0 + k0);
    v16h va1 = frag16(a1 + k0);
    v16h vb0 = frag16(b0 + k0);
    v16h vb1 = frag16(b1 + k0);
    if (k0 + 32 < K){
      __builtin_prefetch(a0 + k0 + 32, 0, 1);   // global_prefetch_b8
      __builtin_prefetch(a1 + k0 + 32, 0, 1);
      __builtin_prefetch(b0 + k0 + 32, 0, 1);
      __builtin_prefetch(b1 + k0 + 32, 0, 1);
    }
    acc00 = wmma_(va0, vb0, acc00);
    acc01 = wmma_(va0, vb1, acc01);
    acc10 = wmma_(va1, vb0, acc10);
    acc11 = wmma_(va1, vb1, acc11);
  }
  // C/D layout per sub-tile: VGPR r -> M = row0 + r + 8*half, N = col0 + (lane&15)
  store_sub(C, ldc, resid, tm*32 +  0 + half*8, tn*32 +  0 + r16, acc00);
  store_sub(C, ldc, resid, tm*32 +  0 + half*8, tn*32 + 16 + r16, acc01);
  store_sub(C, ldc, resid, tm*32 + 16 + half*8, tn*32 +  0 + r16, acc10);
  store_sub(C, ldc, resid, tm*32 + 16 + half*8, tn*32 + 16 + r16, acc11);
}

// ---------------------------------------------------------------------------
// Depthwise causal conv (D_CONV=4) + bias + SiLU.
// Writes f32 (for scan) and f16 (for x_proj WMMA GEMM).
// ---------------------------------------------------------------------------
__global__ void conv_silu_kernel(const float* __restrict__ xz,
                                 const float* __restrict__ cw,
                                 const float* __restrict__ cb,
                                 float* __restrict__ xc,
                                 _Float16* __restrict__ xch){
  int idx = blockIdx.x * blockDim.x + threadIdx.x;   // over NROWS*D_INNER
  int d = idx & (D_INNER - 1);
  int m = idx >> 11;                                 // token row (b*SEQ + l)
  int l = m & (SEQ - 1);
  float acc = cb[d];
#pragma unroll
  for (int k = 0; k < D_CONV; k++){
    int ls = l - (D_CONV - 1) + k;
    if (ls >= 0)
      acc += xz[(size_t)(m - l + ls) * (2*D_INNER) + d] * cw[d*D_CONV + k];
  }
  float r = acc * sigmoidf_(acc);
  xc[idx]  = r;
  xch[idx] = (_Float16)r;
}

// ---------------------------------------------------------------------------
// softplus(dt + bias)
// ---------------------------------------------------------------------------
__global__ void softplus_bias_kernel(float* __restrict__ dtf,
                                     const float* __restrict__ bias){
  int idx = blockIdx.x * blockDim.x + threadIdx.x;
  int d = idx & (D_INNER - 1);
  float v = dtf[idx] + bias[d];
  dtf[idx] = (v > 20.f) ? v : log1pf(__expf(v));
}

// ---------------------------------------------------------------------------
// Selective scan + D skip + SiLU(z) gating, fused; writes f16 y (feeds GEMM).
// One wave = 2 channels x 16 states (lane = s + 16*(d&1)); 2048 waves total.
// ---------------------------------------------------------------------------
__global__ void scan_kernel(const float* __restrict__ dtf,
                            const float* __restrict__ dbc,
                            const float* __restrict__ xc,
                            const float* __restrict__ xz,
                            const float* __restrict__ A_log,
                            const float* __restrict__ Dp,
                            _Float16* __restrict__ yh){
  int gw   = blockIdx.x * (blockDim.x >> 5) + (threadIdx.x >> 5); // 0..2047
  int lane = threadIdx.x & 31;
  int s    = lane & 15;
  int b    = gw >> 10;                  // batch
  int dpr  = gw & 1023;                 // channel pair
  int d    = dpr*2 + (lane >> 4);
  float A  = -__expf(A_log[d*D_STATE + s]);
  float Dv = Dp[d];
  float h  = 0.f;
  for (int t = 0; t < SEQ; t++){
    size_t m = (size_t)b*SEQ + t;
    float dt = dtf[m*D_INNER + d];
    float xv = xc [m*D_INNER + d];
    float Bv = dbc[m*DBC_W + DT_RANK + s];
    float Cv = dbc[m*DBC_W + DT_RANK + D_STATE + s];
    float dA = __expf(dt * A);
    h = dA*h + dt*xv*Bv;
    float p = h * Cv;
    p += __shfl_xor(p, 1, 32);
    p += __shfl_xor(p, 2, 32);
    p += __shfl_xor(p, 4, 32);
    p += __shfl_xor(p, 8, 32);
    if (s == 0){
      float zv = xz[m*(2*D_INNER) + D_INNER + d];
      yh[m*D_INNER + d] = (_Float16)((p + Dv*xv) * (zv * sigmoidf_(zv)));
    }
  }
}

// ---------------------------------------------------------------------------
extern "C" void kernel_launch(void* const* d_in, const int* in_sizes, int n_in,
                              void* d_out, int out_size, void* d_ws, size_t ws_size,
                              hipStream_t stream){
  (void)in_sizes; (void)n_in; (void)out_size; (void)ws_size;
  const float* x    = (const float*)d_in[0];
  const float* lnw  = (const float*)d_in[1];
  const float* lnb  = (const float*)d_in[2];
  const float* Wi   = (const float*)d_in[3];   // (4096, 1024)
  const float* cw   = (const float*)d_in[4];   // (2048, 4)
  const float* cb   = (const float*)d_in[5];
  const float* Wx   = (const float*)d_in[6];   // (96, 2048)
  const float* Wdt  = (const float*)d_in[7];   // (2048, 64)
  const float* bdt  = (const float*)d_in[8];
  const float* Alog = (const float*)d_in[9];
  const float* Dp   = (const float*)d_in[10];
  const float* Wo   = (const float*)d_in[11];  // (1024, 2048)
  float* out = (float*)d_out;

  // --- workspace layout (bytes via float-sized slots) ---
  char* wsb = (char*)d_ws;
  size_t o = 0;
  auto alloc_f32 = [&](size_t n)->float*    { float* p = (float*)(wsb + o);    o += n*4; return p; };
  auto alloc_f16 = [&](size_t n)->_Float16* { _Float16* p = (_Float16*)(wsb + o); o += ((n*2 + 15) & ~(size_t)15); return p; };

  float*    xz   = alloc_f32((size_t)NROWS * 2*D_INNER);   // 64 MB
  float*    xc   = alloc_f32((size_t)NROWS * D_INNER);     // 32 MB
  float*    dbc  = alloc_f32((size_t)NROWS * DBC_W);       // 1.5 MB
  float*    dtf  = alloc_f32((size_t)NROWS * D_INNER);     // 32 MB
  _Float16* xnh  = alloc_f16((size_t)NROWS * D_MODEL);     // 8 MB
  _Float16* Wih  = alloc_f16((size_t)2*D_INNER * D_MODEL); // 8 MB
  _Float16* xch  = alloc_f16((size_t)NROWS * D_INNER);     // 16 MB
  _Float16* Wxh  = alloc_f16((size_t)DBC_W * D_INNER);
  _Float16* Wdth = alloc_f16((size_t)D_INNER * DT_RANK);
  _Float16* dth  = alloc_f16((size_t)NROWS * DT_RANK);
  _Float16* ybh  = alloc_f16((size_t)NROWS * D_INNER);     // 16 MB
  _Float16* Woh  = alloc_f16((size_t)D_MODEL * D_INNER);   // 4 MB

  auto cvt = [&](const float* s, _Float16* dst, int n){
    cvt_f16_kernel<<<(n + 255)/256, 256, 0, stream>>>(s, dst, n);
  };
  auto gemm = [&](const _Float16* A, int lda, const _Float16* B, int ldb,
                  float* C, int ldc, int M, int N, int K, const float* resid){
    int tiles  = (M/32) * (N/32);
    int blocks = (tiles + 3) / 4;          // 4 waves per 128-thread block
    wmma_gemm32<<<blocks, 128, 0, stream>>>(A, lda, B, ldb, C, ldc, M, N, K, resid);
  };

  // one-pass f16 weight conversions (amortized across all K-reuse)
  cvt(Wi,  Wih,  2*D_INNER * D_MODEL);
  cvt(Wx,  Wxh,  DBC_W * D_INNER);
  cvt(Wdt, Wdth, D_INNER * DT_RANK);
  cvt(Wo,  Woh,  D_MODEL * D_INNER);

  ln_kernel<<<NROWS, 256, 0, stream>>>(x, lnw, lnb, xnh);

  // in_proj: xz[4096,4096] = xn * Wi^T
  gemm(xnh, D_MODEL, Wih, D_MODEL, xz, 2*D_INNER, NROWS, 2*D_INNER, D_MODEL, nullptr);
  // depthwise conv + SiLU (f32 + f16 outputs)
  conv_silu_kernel<<<(NROWS*D_INNER)/256, 256, 0, stream>>>(xz, cw, cb, xc, xch);
  // x_proj: dbc[4096,96] = xc * Wx^T
  gemm(xch, D_INNER, Wxh, D_INNER, dbc, DBC_W, NROWS, DBC_W, D_INNER, nullptr);
  // pack dt slice to f16
  dbc_dt_f16_kernel<<<(NROWS*DT_RANK)/256, 256, 0, stream>>>(dbc, dth);
  // dt_proj: dtf[4096,2048] = dt * Wdt^T
  gemm(dth, DT_RANK, Wdth, DT_RANK, dtf, D_INNER, NROWS, D_INNER, DT_RANK, nullptr);
  softplus_bias_kernel<<<(NROWS*D_INNER)/256, 256, 0, stream>>>(dtf, bdt);
  // fused selective scan + D-skip + z gating: 2048 waves = 256 blocks x 8 waves
  scan_kernel<<<(BATCH*D_INNER/2)/8, 256, 0, stream>>>(dtf, dbc, xc, xz, Alog, Dp, ybh);
  // out_proj + residual: out = x + y * Wo^T
  gemm(ybh, D_INNER, Woh, D_INNER, out, D_MODEL, NROWS, D_MODEL, D_INNER, x);
}
